// ScaledDotProductAttention_206158430384
// MI455X (gfx1250) — matrix operand order repaired and checked by reading, not compiled, not run on hardware
//
#include <hip/hip_runtime.h>

typedef _Float16 f16t;
typedef __attribute__((ext_vector_type(16))) _Float16 v16h;
typedef __attribute__((ext_vector_type(8)))  float    v8f;
typedef __attribute__((ext_vector_type(4)))  float    v4f;

constexpr int   N_  = 4, H_ = 16, L_ = 1024, DK_ = 64, DV_ = 64;
constexpr float TEMP_INV = 0.125f;     // 1/sqrt(64)
constexpr float NEG      = -1e9f;
constexpr int   WAVES    = 4;          // waves per block
constexpr int   STRIDE   = L_ + 4;     // LDS row stride (floats), pad vs bank conflicts
constexpr size_t SMEM_BYTES =
    (size_t)(WAVES * 16 * STRIDE + WAVES * 16) * sizeof(float);

__global__ __launch_bounds__(WAVES * 32)
void sdpa_causal_wmma(const float* __restrict__ q, const float* __restrict__ k,
                      const float* __restrict__ v, const float* __restrict__ mask,
                      float* __restrict__ out, float* __restrict__ attn)
{
  extern __shared__ float smem[];
  const int lane = threadIdx.x & 31;
  const int wave = threadIdx.x >> 5;
  const int lo   = lane & 15;   // N / M index within half-wave
  const int hi   = lane >> 4;   // half-wave select
  const int bh   = blockIdx.y;
  const int b    = bh / H_;
  const int r_base = blockIdx.x * (WAVES * 16) + wave * 16;

  const float* Qp = q    + (size_t)bh * L_ * DK_;
  const float* Kp = k    + (size_t)bh * DK_ * L_;   // pre-transposed: (Dk, L)
  const float* Vp = v    + (size_t)bh * L_ * DV_;
  const float* Mp = mask + (size_t)b  * L_ * L_;    // (N,1,L,L) broadcast over H

  float* sm    = smem + wave * 16 * STRIDE;              // 16 x L score/prob buffer
  float* stats = smem + WAVES * 16 * STRIDE + wave * 16; // per-row 1/sum

  // ---- Q A-fragments: 16x32 f16 layout, two K-chunks; fold 1/temperature ----
  v16h aq0, aq1;
  {
    const float* qr = Qp + (size_t)(r_base + lo) * DK_;
    #pragma unroll
    for (int t = 0; t < 8; ++t) {
      aq0[t]     = (f16t)(qr[      hi * 16 + t] * TEMP_INV);
      aq0[8 + t] = (f16t)(qr[8 +   hi * 16 + t] * TEMP_INV);
      aq1[t]     = (f16t)(qr[32 +  hi * 16 + t] * TEMP_INV);
      aq1[8 + t] = (f16t)(qr[40 +  hi * 16 + t] * TEMP_INV);
    }
  }

  const int j_max = (r_base + 15) >> 4;   // last column tile with any unmasked entry
  const int valid = (j_max + 1) * 16;     // = r_base + 16

  // ---- Phase A: S = (Q/T) Kt per 16x16 tile, apply mask, stash in LDS ----
  for (int j = 0; j <= j_max; ++j) {
    const int ncol = j * 16 + lo;
    if (j < j_max) {  // cover latency of next tile's strided K gathers
      __builtin_prefetch(&Kp[(size_t)(hi * 16) * L_ + ncol + 16], 0, 1);
      __builtin_prefetch(&Kp[(size_t)(32 + hi * 16) * L_ + ncol + 16], 0, 1);
    }
    v16h bk0, bk1;                        // 32x16 f16 B layout
    #pragma unroll
    for (int t = 0; t < 16; ++t) {
      bk0[t] = (f16t)Kp[(size_t)(     hi * 16 + t) * L_ + ncol];
      bk1[t] = (f16t)Kp[(size_t)(32 + hi * 16 + t) * L_ + ncol];
    }
    v8f acc = {0.f, 0.f, 0.f, 0.f, 0.f, 0.f, 0.f, 0.f};
    acc = __builtin_amdgcn_wmma_f32_16x16x32_f16(false, aq0, false, bk0,
                                                 (short)0, acc, false, false);
    acc = __builtin_amdgcn_wmma_f32_16x16x32_f16(false, aq1, false, bk1,
                                                 (short)0, acc, false, false);
    #pragma unroll
    for (int i = 0; i < 8; ++i) {         // C/D: VGPR i -> M = i + 8*hi
      const int lr  = i + hi * 8;
      const int row = r_base + lr;
      const float m = Mp[(size_t)row * L_ + (j * 16 + lo)];
      sm[lr * STRIDE + j * 16 + lo] = m * acc[i] + (1.0f - m) * NEG;
    }
  }
  __syncthreads();

  // ---- Phase B: exact row softmax over LDS (store unnormalized exp back) ----
  if (lane < 16) {
    float* rp = sm + lane * STRIDE;
    float mx = -3.402823466e38f;
    for (int c = 0; c < valid; c += 4) {
      v4f x = *(const v4f*)(rp + c);
      mx = fmaxf(mx, fmaxf(fmaxf(x.x, x.y), fmaxf(x.z, x.w)));
    }
    float sum = 0.f;
    for (int c = 0; c < valid; c += 4) {
      v4f x = *(const v4f*)(rp + c);
      v4f p;
      p.x = __expf(x.x - mx); p.y = __expf(x.y - mx);
      p.z = __expf(x.z - mx); p.w = __expf(x.w - mx);
      sum += p.x + p.y + p.z + p.w;
      *(v4f*)(rp + c) = p;
    }
    stats[lane] = 1.0f / sum;
  }
  __syncthreads();

  // ---- Phase C1: stream normalized attn rows to HBM exactly once (NT) ----
  for (int lr = 0; lr < 16; ++lr) {
    const int   row = r_base + lr;
    const float inv = stats[lr];
    float*       dst = attn + ((size_t)bh * L_ + row) * L_;
    const float* src = sm + lr * STRIDE;
    for (int c4 = lane * 4; c4 < L_; c4 += 32 * 4) {
      v4f val;
      if (c4 < valid) {
        v4f x = *(const v4f*)(src + c4);
        val.x = x.x * inv; val.y = x.y * inv; val.z = x.z * inv; val.w = x.w * inv;
      } else {
        val = (v4f){0.f, 0.f, 0.f, 0.f};  // strictly-upper causal region
      }
      __builtin_nontemporal_store(val, (v4f*)(dst + c4));  // write-once: keep L2
    }
  }

  // ---- Phase C2: O = P V via WMMA (unnormalized P from LDS; scale rows last) ----
  v8f acc0 = {0.f,0.f,0.f,0.f,0.f,0.f,0.f,0.f};
  v8f acc1 = acc0, acc2 = acc0, acc3 = acc0;
  const int nchunks = (valid + 31) >> 5;
  for (int mc = 0; mc < nchunks; ++mc) {
    const int kb = mc * 32;
    if (mc + 1 < nchunks)   // prefetch next V chunk rows
      __builtin_prefetch(&Vp[(size_t)(kb + 32 + hi * 16) * DV_ + lo], 0, 1);
    v16h ap;                               // P tile 16x32 as A fragment
    #pragma unroll
    for (int t = 0; t < 8; ++t) {
      const int k0 = kb +      hi * 16 + t;
      const int k1 = kb + 8 +  hi * 16 + t;
      ap[t]     = (k0 < valid) ? (f16t)sm[lo * STRIDE + k0] : (f16t)0.f;
      ap[8 + t] = (k1 < valid) ? (f16t)sm[lo * STRIDE + k1] : (f16t)0.f;
    }
    v16h bv;
    #pragma unroll
    for (int t = 0; t < 16; ++t)
      bv[t] = (f16t)Vp[(size_t)(kb + hi * 16 + t) * DV_ +  0 + lo];
    acc0 = __builtin_amdgcn_wmma_f32_16x16x32_f16(false, ap, false, bv, (short)0, acc0, false, false);
    #pragma unroll
    for (int t = 0; t < 16; ++t)
      bv[t] = (f16t)Vp[(size_t)(kb + hi * 16 + t) * DV_ + 16 + lo];
    acc1 = __builtin_amdgcn_wmma_f32_16x16x32_f16(false, ap, false, bv, (short)0, acc1, false, false);
    #pragma unroll
    for (int t = 0; t < 16; ++t)
      bv[t] = (f16t)Vp[(size_t)(kb + hi * 16 + t) * DV_ + 32 + lo];
    acc2 = __builtin_amdgcn_wmma_f32_16x16x32_f16(false, ap, false, bv, (short)0, acc2, false, false);
    #pragma unroll
    for (int t = 0; t < 16; ++t)
      bv[t] = (f16t)Vp[(size_t)(kb + hi * 16 + t) * DV_ + 48 + lo];
    acc3 = __builtin_amdgcn_wmma_f32_16x16x32_f16(false, ap, false, bv, (short)0, acc3, false, false);
  }

  #pragma unroll
  for (int i = 0; i < 8; ++i) {
    const int lr  = i + hi * 8;
    const int row = r_base + lr;
    const float inv = stats[lr];
    float* orow = out + ((size_t)bh * L_ + row) * DV_;
    __builtin_nontemporal_store(acc0[i] * inv, orow +  0 + lo);
    __builtin_nontemporal_store(acc1[i] * inv, orow + 16 + lo);
    __builtin_nontemporal_store(acc2[i] * inv, orow + 32 + lo);
    __builtin_nontemporal_store(acc3[i] * inv, orow + 48 + lo);
  }
}

extern "C" void kernel_launch(void* const* d_in, const int* in_sizes, int n_in,
                              void* d_out, int out_size, void* d_ws, size_t ws_size,
                              hipStream_t stream) {
  (void)in_sizes; (void)n_in; (void)out_size; (void)d_ws; (void)ws_size;
  const float* q    = (const float*)d_in[0];
  const float* k    = (const float*)d_in[1];
  const float* v    = (const float*)d_in[2];
  const float* mask = (const float*)d_in[3];
  float* out  = (float*)d_out;                                // (N,H,L,DV)
  float* attn = out + (size_t)N_ * H_ * L_ * DV_;             // (N,H,L,L)

  (void)hipFuncSetAttribute((const void*)sdpa_causal_wmma,
                            hipFuncAttributeMaxDynamicSharedMemorySize,
                            (int)SMEM_BYTES);
  dim3 grid(L_ / (WAVES * 16), N_ * H_);
  sdpa_causal_wmma<<<grid, WAVES * 32, SMEM_BYTES, stream>>>(q, k, v, mask,
                                                             out, attn);
}
